// GNS_83906481094854
// MI455X (gfx1250) — matrix operand order repaired and checked by reading, not compiled, not run on hardware
//
#include <hip/hip_runtime.h>

#define HID 128
#define N_NODES 40000
#define N_EDGES 640000
#define ROWS 64            // rows per block: 4 M-tiles; each wave owns 2M x 2N tiles

typedef __attribute__((ext_vector_type(16))) __bf16 v16bf;
typedef __attribute__((ext_vector_type(8)))  __bf16 v8bf;
typedef __attribute__((ext_vector_type(8)))  float  v8f;

struct Mlp4 {
  const __bf16* w[4];
  const float*  b[4];
  const float*  g;
  const float*  be;
};

struct LaunchArgs {
  Mlp4 mlp;
  const float* in0;   // direct input (enc/dec) or agg (upd)
  const float* in1;   // lv (msg gather / upd concat)
  const float* in2;   // le (msg)
  const int* src;
  const int* dst;
  float* out;         // lv / le / d_out
  float* agg;         // scatter target (msg)
  int nrows;
  int K0;
  int mode;           // 0 enc, 1 msg, 2 upd, 3 dec
};

// Pack f32 weight matrix W[K][N] into per-lane WMMA bf16 B-fragments.
// Fragment (kb, nb): lane = khalf*16 + n ; value v -> K = kb*32 + khalf*8 + (v&7) + (v>>3)*16.
__global__ void pack_w_kernel(const float* __restrict__ W, int K, int N,
                              __bf16* __restrict__ dstp) {
  int nnb  = (N + 15) >> 4;
  int frag = blockIdx.x;
  int kb = frag / nnb, nb = frag - kb * nnb;
  int lane  = threadIdx.x;
  int n     = nb * 16 + (lane & 15);
  int khalf = lane >> 4;
  __bf16* d = dstp + ((size_t)frag * 32 + lane) * 16;
#pragma unroll
  for (int v = 0; v < 16; ++v) {
    int k = kb * 32 + khalf * 8 + (v & 7) + (v >> 3) * 16;
    float x = (k < K && n < N) ? W[(size_t)k * N + n] : 0.0f;
    d[v] = (__bf16)x;
  }
}

#define SPAD 392   // 16B-aligned bf16 row stride (784 B), bank-conflict padded

__device__ __forceinline__ v16bf load_a_frag(const __bf16 (*s_in)[SPAD],
                                             int rowA, int kb, int half) {
  const __bf16* ap = &s_in[rowA][kb * 32 + half * 8];
  v8bf lo = *(const v8bf*)(ap);        // K = kb*32 + half*8 .. +7
  v8bf hi = *(const v8bf*)(ap + 16);   // K = kb*32 + half*8 + 16 .. +23
  v16bf a;
#pragma unroll
  for (int i = 0; i < 8; ++i) { a[i] = lo[i]; a[i + 8] = hi[i]; }
  return a;
}

// f32 view of s_in[r][128..383]: dead after layer 0, reused for final layer output
__device__ __forceinline__ float* sout_row(__bf16 (*s_in)[SPAD], int r) {
  return (float*)((char*)&s_in[r][0] + 256);
}

__global__ __launch_bounds__(256) void fused_mlp_kernel(LaunchArgs a) {
  __shared__ __bf16 s_in[ROWS][SPAD];     // 50 KB
  __shared__ float  s_red[ROWS][8];       // LN partials (sum, sumsq)
  __shared__ float  s_mu[ROWS];
  __shared__ float  s_rs[ROWS];

  const int tid   = threadIdx.x;
  const int wave  = tid >> 5;
  const int lane  = tid & 31;
  const int rbase = blockIdx.x * ROWS;

  const int K0 = a.K0;
  const int Kp = (K0 + 31) & ~31;

  // ---- stage ROWS x Kp input tile into LDS as bf16 (fused gather/concat) ----
  if ((K0 & 3) == 0) {
    // vectorized: float4 global loads, packed 4x bf16 LDS store (ds_store_b64)
    const int nch = Kp >> 2;
    for (int idx = tid; idx < ROWS * nch; idx += 256) {
      int r = idx / nch, c4 = (idx - r * nch) * 4;
      int row = rbase + r;
      float4 v = make_float4(0.f, 0.f, 0.f, 0.f);
      if (row < a.nrows && c4 + 3 < K0) {
        const float* sp;
        if (a.mode == 1) {          // concat(lv[dst], lv[src], le)
          if (c4 < HID)          sp = &a.in1[(size_t)a.dst[row] * HID + c4];
          else if (c4 < 2 * HID) sp = &a.in1[(size_t)a.src[row] * HID + (c4 - HID)];
          else                   sp = &a.in2[(size_t)row * HID + (c4 - 2 * HID)];
        } else if (a.mode == 2) {   // concat(agg, lv)
          if (c4 < HID) sp = &a.in0[(size_t)row * HID + c4];
          else          sp = &a.in1[(size_t)row * HID + (c4 - HID)];
        } else {
          sp = &a.in0[(size_t)row * K0 + c4];
        }
        v = *(const float4*)sp;
      }
      union { __bf16 h[4]; uint2 u; } pk;
      pk.h[0] = (__bf16)v.x; pk.h[1] = (__bf16)v.y;
      pk.h[2] = (__bf16)v.z; pk.h[3] = (__bf16)v.w;
      *(uint2*)&s_in[r][c4] = pk.u;
    }
  } else {
    // scalar fallback (enc_V, K0=30)
    for (int idx = tid; idx < ROWS * Kp; idx += 256) {
      int r = idx / Kp, c = idx - r * Kp;
      int row = rbase + r;
      float v = (row < a.nrows && c < K0) ? a.in0[(size_t)row * K0 + c] : 0.0f;
      s_in[r][c] = (__bf16)v;
    }
  }
  __syncthreads();

  // tile assignment: wave -> M tiles {mb, mb+1}, N tiles {nb, nb+1}
  const int mb = (wave & 1) * 2;
  const int nb = (wave >> 1) * 2;
  const int half = lane >> 4;
  const int rowA = lane & 15;

  // ---- 4 fused Linear layers (LeakyReLU on first 3) ----
  for (int l = 0; l < 4; ++l) {
    const int Kl   = (l == 0) ? Kp : HID;
    const int nnb  = (a.mode == 3 && l == 3) ? 1 : 8;
    const int Nact = (a.mode == 3 && l == 3) ? 3 : HID;
    const bool n0 = (nb < nnb);
    const bool n1 = (nb + 1 < nnb);
    v8f acc00 = {}, acc01 = {}, acc10 = {}, acc11 = {};
    if (n0) {
      const int nkb = Kl >> 5;
      const __bf16* wp = a.mlp.w[l];
      __builtin_prefetch(wp + (size_t)nb * 512, 0, 1);   // global_prefetch_b8
      for (int kb = 0; kb < nkb; ++kb) {
        v16bf a0 = load_a_frag(s_in, rowA + 16 * mb,       kb, half);
        v16bf a1 = load_a_frag(s_in, rowA + 16 * (mb + 1), kb, half);
        v16bf b0 = *(const v16bf*)(wp + ((size_t)(kb * nnb + nb) * 32 + lane) * 16);
        acc00 = __builtin_amdgcn_wmma_f32_16x16x32_bf16(false, a0, false, b0,
                                                        (short)0, acc00, false, false);
        acc10 = __builtin_amdgcn_wmma_f32_16x16x32_bf16(false, a1, false, b0,
                                                        (short)0, acc10, false, false);
        if (n1) {
          v16bf b1 = *(const v16bf*)(wp + ((size_t)(kb * nnb + nb + 1) * 32 + lane) * 16);
          acc01 = __builtin_amdgcn_wmma_f32_16x16x32_bf16(false, a0, false, b1,
                                                          (short)0, acc01, false, false);
          acc11 = __builtin_amdgcn_wmma_f32_16x16x32_bf16(false, a1, false, b1,
                                                          (short)0, acc11, false, false);
        }
      }
      const int col0 = nb * 16 + (lane & 15);
      const int col1 = col0 + 16;
      const float bv0 = (col0 < Nact) ? a.mlp.b[l][col0] : 0.0f;
      const float bv1 = (n1 && col1 < Nact) ? a.mlp.b[l][col1] : 0.0f;
#pragma unroll
      for (int j = 0; j < 8; ++j) {
        float x00 = acc00[j] + bv0, x10 = acc10[j] + bv0;
        float x01 = acc01[j] + bv1, x11 = acc11[j] + bv1;
        if (l < 3) {
          x00 = (x00 > 0.f) ? x00 : 0.01f * x00;
          x10 = (x10 > 0.f) ? x10 : 0.01f * x10;
          x01 = (x01 > 0.f) ? x01 : 0.01f * x01;
          x11 = (x11 > 0.f) ? x11 : 0.01f * x11;
        }
        acc00[j] = x00; acc10[j] = x10; acc01[j] = x01; acc11[j] = x11;
      }
    }
    __syncthreads();   // all reads of s_in done before overwrite
    if (n0) {
      const int col0 = nb * 16 + (lane & 15);
      const int col1 = col0 + 16;
      const int r0   = (lane >> 4) * 8;         // C/D layout: M = j + 8*(lane/16)
      const int ra   = 16 * mb + r0;
      const int rb   = 16 * (mb + 1) + r0;
      if (l < 3) {
#pragma unroll
        for (int j = 0; j < 8; ++j) {
          s_in[ra + j][col0] = (__bf16)acc00[j];
          s_in[rb + j][col0] = (__bf16)acc10[j];
          if (n1) {
            s_in[ra + j][col1] = (__bf16)acc01[j];
            s_in[rb + j][col1] = (__bf16)acc11[j];
          }
        }
      } else {
#pragma unroll
        for (int j = 0; j < 8; ++j) {
          sout_row(s_in, ra + j)[col0] = acc00[j];
          sout_row(s_in, rb + j)[col0] = acc10[j];
          if (n1) {
            sout_row(s_in, ra + j)[col1] = acc01[j];
            sout_row(s_in, rb + j)[col1] = acc11[j];
          }
        }
      }
    }
    __syncthreads();
  }

  // ---- LayerNorm over 128 features: 4 threads/row partial sum & sumsq ----
  if (a.mode != 3) {
    {
      int r = tid >> 2, q = tid & 3;
      const float* orow = sout_row(s_in, r);
      float s = 0.f, s2 = 0.f;
      for (int c = q * 32; c < q * 32 + 32; c += 4) {
        float4 x = *(const float4*)(orow + c);
        s  += x.x + x.y + x.z + x.w;
        s2 += x.x * x.x + x.y * x.y + x.z * x.z + x.w * x.w;
      }
      s_red[r][q] = s;
      s_red[r][q + 4] = s2;
    }
    __syncthreads();
    if (tid < ROWS) {
      float s  = s_red[tid][0] + s_red[tid][1] + s_red[tid][2] + s_red[tid][3];
      float s2 = s_red[tid][4] + s_red[tid][5] + s_red[tid][6] + s_red[tid][7];
      float mu  = s * (1.0f / HID);
      float var = s2 * (1.0f / HID) - mu * mu;
      s_mu[tid] = mu;
      s_rs[tid] = rsqrtf(fmaxf(var, 0.0f) + 1e-5f);
    }
    __syncthreads();
  }

  // ---- epilogue: LN affine + residual / scatter-add / decode store ----
  if (a.mode == 3) {
    for (int idx = tid; idx < ROWS * 3; idx += 256) {
      int r = idx / 3, c = idx - r * 3;
      int row = rbase + r;
      if (row < a.nrows) a.out[(size_t)row * 3 + c] = sout_row(s_in, r)[c];
    }
  } else {
    for (int idx = tid; idx < ROWS * 32; idx += 256) {   // float4 chunks
      int r = idx >> 5, c4 = (idx & 31) * 4;
      int row = rbase + r;
      if (row >= a.nrows) continue;
      float4 y = *(const float4*)(sout_row(s_in, r) + c4);
      const float mu = s_mu[r], rs = s_rs[r];
      float4 g  = *(const float4*)(a.mlp.g  + c4);
      float4 be = *(const float4*)(a.mlp.be + c4);
      y.x = (y.x - mu) * rs * g.x + be.x;
      y.y = (y.y - mu) * rs * g.y + be.y;
      y.z = (y.z - mu) * rs * g.z + be.z;
      y.w = (y.w - mu) * rs * g.w + be.w;
      float* op = &a.out[(size_t)row * HID + c4];
      if (a.mode == 0) {
        *(float4*)op = y;                                    // lv or le
      } else if (a.mode == 1) {
        float4 e = *(const float4*)(&a.in2[(size_t)row * HID + c4]);
        e.x += y.x; e.y += y.y; e.z += y.z; e.w += y.w;
        *(float4*)op = e;                                    // le += msg
        float* ap = &a.agg[(size_t)a.dst[row] * HID + c4];   // segment_sum(msg,dst)
        atomicAdd(ap + 0, y.x);
        atomicAdd(ap + 1, y.y);
        atomicAdd(ap + 2, y.z);
        atomicAdd(ap + 3, y.w);
      } else {                                               // lv += upd
        float4 v = *(const float4*)op;
        v.x += y.x; v.y += y.y; v.z += y.z; v.w += y.w;
        *(float4*)op = v;
      }
    }
  }
}

extern "C" void kernel_launch(void* const* d_in, const int* in_sizes, int n_in,
                              void* d_out, int out_size, void* d_ws, size_t ws_size,
                              hipStream_t stream) {
  const float* V  = (const float*)d_in[0];
  const float* E  = (const float*)d_in[1];
  const int*   ei = (const int*)d_in[2];
  const int* esrc = ei;              // edge_index[0]
  const int* edst = ei + N_EDGES;    // edge_index[1]

  char* ws = (char*)d_ws;
  size_t o = 0;
  float* lv  = (float*)(ws + o); o += (size_t)N_NODES * HID * sizeof(float);
  float* le  = (float*)(ws + o); o += (size_t)N_EDGES * HID * sizeof(float);
  float* agg = (float*)(ws + o); o += (size_t)N_NODES * HID * sizeof(float);

  auto packW = [&](int inIdx, int K, int N) -> const __bf16* {
    int nkb = (K + 31) / 32, nnb = (N + 15) / 16;
    __bf16* dstp = (__bf16*)(ws + o);
    o += (size_t)nkb * nnb * 512 * sizeof(__bf16);
    pack_w_kernel<<<dim3(nkb * nnb), dim3(32), 0, stream>>>(
        (const float*)d_in[inIdx], K, N, dstp);
    return dstp;
  };

  auto buildMlp = [&](int base, int K0, int Nlast, bool ln) -> Mlp4 {
    Mlp4 m;
    int K = K0;
    for (int l = 0; l < 4; ++l) {
      int N = (l == 3) ? Nlast : HID;
      m.w[l] = packW(base + 2 * l, K, N);
      m.b[l] = (const float*)d_in[base + 2 * l + 1];
      K = N;
    }
    m.g  = ln ? (const float*)d_in[base + 8] : nullptr;
    m.be = ln ? (const float*)d_in[base + 9] : nullptr;
    return m;
  };

  // d_in layout (setup_inputs insertion order): V, E, edge_index, then params:
  // enc_V(10), enc_E(10), p1.msg(10), p1.upd(10), p2.msg, p2.upd, p3.msg, p3.upd, dec(8)
  Mlp4 encV = buildMlp(3, 30, HID, true);
  Mlp4 encE = buildMlp(13, 4, HID, true);
  Mlp4 msgM[3] = { buildMlp(23, 3 * HID, HID, true),
                   buildMlp(43, 3 * HID, HID, true),
                   buildMlp(63, 3 * HID, HID, true) };
  Mlp4 updM[3] = { buildMlp(33, 2 * HID, HID, true),
                   buildMlp(53, 2 * HID, HID, true),
                   buildMlp(73, 2 * HID, HID, true) };
  Mlp4 decM = buildMlp(83, HID, 3, false);

  LaunchArgs a;

  a = {}; a.mlp = encV; a.in0 = V; a.out = lv; a.nrows = N_NODES; a.K0 = 30; a.mode = 0;
  fused_mlp_kernel<<<N_NODES / ROWS, 256, 0, stream>>>(a);

  a = {}; a.mlp = encE; a.in0 = E; a.out = le; a.nrows = N_EDGES; a.K0 = 4; a.mode = 0;
  fused_mlp_kernel<<<N_EDGES / ROWS, 256, 0, stream>>>(a);

  for (int p = 0; p < 3; ++p) {
    hipMemsetAsync(agg, 0, (size_t)N_NODES * HID * sizeof(float), stream);

    a = {}; a.mlp = msgM[p]; a.in1 = lv; a.in2 = le; a.src = esrc; a.dst = edst;
    a.out = le; a.agg = agg; a.nrows = N_EDGES; a.K0 = 3 * HID; a.mode = 1;
    fused_mlp_kernel<<<N_EDGES / ROWS, 256, 0, stream>>>(a);

    a = {}; a.mlp = updM[p]; a.in0 = agg; a.in1 = lv; a.out = lv;
    a.nrows = N_NODES; a.K0 = 2 * HID; a.mode = 2;
    fused_mlp_kernel<<<N_NODES / ROWS, 256, 0, stream>>>(a);
  }

  a = {}; a.mlp = decM; a.in0 = lv; a.out = (float*)d_out;
  a.nrows = N_NODES; a.K0 = HID; a.mode = 3;
  fused_mlp_kernel<<<N_NODES / ROWS, 256, 0, stream>>>(a);
}
